// MASK_EGNN_NET_85383949845227
// MI455X (gfx1250) — matrix operand order repaired and checked by reading, not compiled, not run on hardware
//
#include <hip/hip_runtime.h>
#include <hip/hip_fp16.h>

typedef __attribute__((ext_vector_type(16))) _Float16 v16h;
typedef __attribute__((ext_vector_type(8)))  float    v8f;

#define N_NODES 50000
#define N_EDGES 600000
#define FEATD 64
#define MDIMD 64
#define EDIMD 32
#define EIND  161      // 2*FEAT + EDIM + 1
#define EHID  322
#define CHID  256
#define OUTD  20
#define TE 32          // edges per block
#define TN 64          // node rows per block

// fast silu: x * sigmoid(x) with hardware v_rcp_f32 (avoids IEEE div sequence)
__device__ __forceinline__ float silu_f(float x) {
    return x * __builtin_amdgcn_rcpf(1.0f + __expf(-x));
}

// ---- WMMA helpers (16x16x32 f16 -> f32) ----
__device__ __forceinline__ v8f wmma_f16(v16h a, v16h b, v8f c) {
    return __builtin_amdgcn_wmma_f32_16x16x32_f16(
        /*neg_a=*/false, a, /*neg_b=*/false, b,
        /*c_mod=*/(short)0, c, /*reuse_a=*/false, /*reuse_b=*/false);
}

// A fragment (16 rows x 32 K, f16) from LDS tile; `base` = tile origin, `stride` in elements.
// Per ISA layout: lane L holds row M=L%16; elements e<8 -> K = e + 8*(L/16),
// elements e>=8 -> K = 16 + (e-8) + 8*(L/16).  (merges into 2x ds_load_b128)
__device__ __forceinline__ v16h load_A_lds(const _Float16* base, int stride, int lane, int kbase) {
    const _Float16* row = base + (lane & 15) * stride + kbase + ((lane >> 4) << 3);
    v16h a;
#pragma unroll
    for (int e = 0; e < 16; ++e) {
        int k = (e & 7) + ((e >> 3) << 4);
        a[e] = row[k];
    }
    return a;
}

// B fragment from pre-packed per-lane layout: one aligned 32B vector load.
__device__ __forceinline__ v16h load_B_pk(const _Float16* base, int NT, int lane, int kc, int nt) {
    const v16h* p = (const v16h*)base + ((kc * NT + nt) * 32 + lane);
    return *p;
}

// =====================================================================
// pack weights W[K x N] (f32, row-major) into WMMA B-fragment order (f16):
// out[(((kc*NT)+nt)*32 + lane)*16 + e] = W[k,n], k = kc*32+8*(lane/16)+(e%8)+16*(e/8),
// n = nt*16 + lane%16, zero-padded outside K x N.
// =====================================================================
__global__ void pack_w_kernel(const float* __restrict__ W, _Float16* __restrict__ out,
                              int K, int N, int KC, int NT) {
    int total = KC * NT * 512;
    for (int i = blockIdx.x * blockDim.x + threadIdx.x; i < total; i += gridDim.x * blockDim.x) {
        int e = i & 15;
        int lane = (i >> 4) & 31;
        int t = i >> 9;
        int nt = t % NT, kc = t / NT;
        int n = nt * 16 + (lane & 15);
        int k = kc * 32 + ((lane >> 4) << 3) + (e & 7) + ((e >> 3) << 4);
        float v = (k < K && n < N) ? W[(size_t)k * N + n] : 0.0f;
        out[i] = (_Float16)v;
    }
}

// =====================================================================
// prep: feats = concat(x, extra_x); coors = pos
// =====================================================================
__global__ void prep_kernel(const float* __restrict__ x, const float* __restrict__ extra_x,
                            const float* __restrict__ pos,
                            float* __restrict__ feats, float* __restrict__ coors) {
    int i = blockIdx.x * blockDim.x + threadIdx.x;
    if (i < N_NODES * 64) {
        int row = i >> 6, c = i & 63;
        feats[i] = (c < 48) ? x[row * 48 + c] : extra_x[row * 16 + (c - 48)];
    }
    if (i < N_NODES * 3) coors[i] = pos[i];
}

// =====================================================================
// ss embedding: ss_embed = W2 @ silu(W1 @ ss + b1) + b2
// =====================================================================
__global__ void __launch_bounds__(256)
ss_kernel(const float* __restrict__ ss,
          const _Float16* __restrict__ pW1, const float* __restrict__ b1,
          const _Float16* __restrict__ pW2, const float* __restrict__ b2,
          float* __restrict__ ss_embed) {
    __shared__ __align__(16) _Float16 sh_in[TN * 32];
    __shared__ __align__(16) _Float16 sh_h[TN * 64];
    int tid = threadIdx.x, lane = tid & 31, w = tid >> 5;
    int r0 = blockIdx.x * TN;

    for (int i = tid; i < TN * 32; i += 256) {
        int row = i >> 5, c = i & 31;
        int g = r0 + row;
        float v = (g < N_NODES && c < 8) ? ss[(size_t)g * 8 + c] : 0.0f;
        sh_in[i] = (_Float16)v;
    }
    __syncthreads();
    {   // layer1: K=8 (one padded chunk), N=64
        const _Float16* Ab = sh_in + (w & 3) * 16 * 32;
        for (int nt = (w >> 2); nt < 4; nt += 2) {
            v8f acc = {};
            v16h a = load_A_lds(Ab, 32, lane, 0);
            v16h b = load_B_pk(pW1, 4, lane, 0, nt);
            acc = wmma_f16(a, b, acc);
            int n = nt * 16 + (lane & 15);
            float bias = b1[n];
            int mb = (w & 3) * 16 + ((lane >> 4) << 3);
#pragma unroll
            for (int r = 0; r < 8; ++r)
                sh_h[(mb + r) * 64 + n] = (_Float16)silu_f(acc[r] + bias);
        }
    }
    __syncthreads();
    {   // layer2: K=64, N=64
        const _Float16* Ab = sh_h + (w & 3) * 16 * 64;
        for (int nt = (w >> 2); nt < 4; nt += 2) {
            v8f acc = {};
#pragma unroll
            for (int kc = 0; kc < 2; ++kc) {
                v16h a = load_A_lds(Ab, 64, lane, kc * 32);
                v16h b = load_B_pk(pW2, 4, lane, kc, nt);
                acc = wmma_f16(a, b, acc);
            }
            int n = nt * 16 + (lane & 15);
            float bias = b2[n];
            int mb = (w & 3) * 16 + ((lane >> 4) << 3);
#pragma unroll
            for (int r = 0; r < 8; ++r) {
                int g = r0 + mb + r;
                if (g < N_NODES) ss_embed[(size_t)g * 64 + n] = acc[r] + bias;
            }
        }
    }
}

// =====================================================================
// Fused edge pipeline: e_in -> h1(322) -> m(64) -> {ea(32), c1(256) -> w}
// segment sums via atomics into m_i / cdelta.
// =====================================================================
__global__ void __launch_bounds__(256)
edge_kernel(const float* __restrict__ feats, const float* __restrict__ coors,
            const int* __restrict__ eidx,
            const float* __restrict__ ea_f32, const _Float16* __restrict__ ea_f16,
            const _Float16* __restrict__ pW1, const float* __restrict__ b1,
            const _Float16* __restrict__ pW2, const float* __restrict__ b2,
            const _Float16* __restrict__ pWc1, const float* __restrict__ bc1,
            const float* __restrict__ Wc2, const float* __restrict__ bc2,
            const _Float16* __restrict__ pWeo, const float* __restrict__ beo,
            float* __restrict__ m_i, float* __restrict__ cdelta,
            _Float16* __restrict__ ea_out, int write_ea) {
    __shared__ __align__(16) _Float16 sh_ein[TE * 192];  // e_in, K padded 161->192
    __shared__ __align__(16) _Float16 sh_h1[TE * 352];   // h1 (322->352); reused as c1 (stride 256)
    __shared__ __align__(16) _Float16 sh_m[TE * 64];
    __shared__ float sh_rel[TE * 3];
    __shared__ int sh_src[TE];
    __shared__ int sh_dst[TE];

    const int tid = threadIdx.x;
    const int lane = tid & 31;
    const int w = tid >> 5;
    const int e0 = blockIdx.x * TE;

    for (int i = tid; i < TE * 352; i += 256) sh_h1[i] = (_Float16)0.0f;

    if (tid < TE) {
        int e = e0 + tid;
        int s = eidx[e];
        int d = eidx[N_EDGES + e];
        sh_src[tid] = s; sh_dst[tid] = d;
        float dx = coors[s * 3 + 0] - coors[d * 3 + 0];
        float dy = coors[s * 3 + 1] - coors[d * 3 + 1];
        float dz = coors[s * 3 + 2] - coors[d * 3 + 2];
        sh_rel[tid * 3 + 0] = dx; sh_rel[tid * 3 + 1] = dy; sh_rel[tid * 3 + 2] = dz;
    }
    __syncthreads();

    // build e_in = [feats[dst], feats[src], ea, dist, 0-pad]
    for (int i = tid; i < TE * 192; i += 256) {
        int e = i / 192, c = i - e * 192;
        float v;
        if (c < 64)       v = feats[(size_t)sh_dst[e] * 64 + c];
        else if (c < 128) v = feats[(size_t)sh_src[e] * 64 + (c - 64)];
        else if (c < 160) v = ea_f32 ? ea_f32[(size_t)(e0 + e) * 32 + (c - 128)]
                                     : (float)ea_f16[(size_t)(e0 + e) * 32 + (c - 128)];
        else if (c == 160) {
            const float* r = &sh_rel[e * 3];
            v = r[0] * r[0] + r[1] * r[1] + r[2] * r[2];
        } else v = 0.0f;
        sh_ein[i] = (_Float16)v;
    }
    __syncthreads();

    // Phase 1: h1 = silu(e_in @ W1 + b1)   K=192(6 chunks), N=322 (21 tiles)
    {
        const _Float16* Ab = sh_ein + (w & 1) * 16 * 192;
        _Float16* Hb = sh_h1 + (w & 1) * 16 * 352;
        for (int nt = (w >> 1); nt < 21; nt += 4) {
            v8f acc = {};
#pragma unroll
            for (int kc = 0; kc < 6; ++kc) {
                v16h a = load_A_lds(Ab, 192, lane, kc * 32);
                v16h b = load_B_pk(pW1, 21, lane, kc, nt);
                acc = wmma_f16(a, b, acc);
            }
            int n = nt * 16 + (lane & 15);
            if (n < EHID) {
                float bias = b1[n];
                int mr = ((lane >> 4) << 3);
#pragma unroll
                for (int r = 0; r < 8; ++r)
                    Hb[(mr + r) * 352 + n] = (_Float16)silu_f(acc[r] + bias);
            }
        }
    }
    __syncthreads();

    // Phase 2: m = silu(h1 @ W2 + b2)   K=352(11), N=64 (4 tiles, one per wave pair)
    {
        const _Float16* Ab = sh_h1 + (w & 1) * 16 * 352;
        int nt = w >> 1;
        v8f acc = {};
#pragma unroll
        for (int kc = 0; kc < 11; ++kc) {
            v16h a = load_A_lds(Ab, 352, lane, kc * 32);
            v16h b = load_B_pk(pW2, 4, lane, kc, nt);
            acc = wmma_f16(a, b, acc);
        }
        int n = nt * 16 + (lane & 15);
        float bias = b2[n];
        int mb = (w & 1) * 16 + ((lane >> 4) << 3);
#pragma unroll
        for (int r = 0; r < 8; ++r) {
            float v = silu_f(acc[r] + bias);
            sh_m[(mb + r) * 64 + n] = (_Float16)v;
            atomicAdd(&m_i[(size_t)sh_dst[mb + r] * 64 + n], v);  // segment_sum(m, dst)
        }
    }
    __syncthreads();

    // Phase ea (not last layer): ea_out = m @ Weo + beo   K=64(2), N=32 (2 tiles)
    if (write_ea && w < 4) {
        const _Float16* Ab = sh_m + (w & 1) * 16 * 64;
        int nt = w >> 1;
        v8f acc = {};
#pragma unroll
        for (int kc = 0; kc < 2; ++kc) {
            v16h a = load_A_lds(Ab, 64, lane, kc * 32);
            v16h b = load_B_pk(pWeo, 2, lane, kc, nt);
            acc = wmma_f16(a, b, acc);
        }
        int n = nt * 16 + (lane & 15);
        float bias = beo[n];
        int mb = (w & 1) * 16 + ((lane >> 4) << 3);
#pragma unroll
        for (int r = 0; r < 8; ++r)
            ea_out[(size_t)(e0 + mb + r) * 32 + n] = (_Float16)(acc[r] + bias);
    }

    // Phase 3: c1 = silu(m @ Wc1 + bc1)  K=64(2), N=256 (16 tiles), overlay on sh_h1
    {
        const _Float16* Ab = sh_m + (w & 1) * 16 * 64;
        for (int nt = (w >> 1); nt < 16; nt += 4) {
            v8f acc = {};
#pragma unroll
            for (int kc = 0; kc < 2; ++kc) {
                v16h a = load_A_lds(Ab, 64, lane, kc * 32);
                v16h b = load_B_pk(pWc1, 16, lane, kc, nt);
                acc = wmma_f16(a, b, acc);
            }
            int n = nt * 16 + (lane & 15);
            float bias = bc1[n];
            int mb = (w & 1) * 16 + ((lane >> 4) << 3);
#pragma unroll
            for (int r = 0; r < 8; ++r)
                sh_h1[(mb + r) * 256 + n] = (_Float16)silu_f(acc[r] + bias);
        }
    }
    __syncthreads();

    // Phase 4: w = c1 @ Wc2 + bc2 ; coors delta scatter
    if (tid < TE) {
        const _Float16* crow = sh_h1 + tid * 256;
        float sum = bc2[0];
#pragma unroll 8
        for (int j = 0; j < 256; ++j) sum += (float)crow[j] * Wc2[j];
        const float* rel = &sh_rel[tid * 3];
        int d = sh_dst[tid];
        atomicAdd(&cdelta[d * 3 + 0], sum * rel[0]);
        atomicAdd(&cdelta[d * 3 + 1], sum * rel[1]);
        atomicAdd(&cdelta[d * 3 + 2], sum * rel[2]);
    }
}

// =====================================================================
// node1: commit coors delta; t = feats + node2(silu(node1(cat(feats,m_i))));
//        u = silu(ff1(t)); accumulate LN stats.
// =====================================================================
__global__ void __launch_bounds__(256)
node1_kernel(float* __restrict__ feats, const float* __restrict__ m_i,
             const float* __restrict__ cdelta, float* __restrict__ coors,
             const _Float16* __restrict__ pWn1, const float* __restrict__ bn1,
             const _Float16* __restrict__ pWn2, const float* __restrict__ bn2,
             const _Float16* __restrict__ pWf1, const float* __restrict__ bf1,
             float* __restrict__ u_buf, float* __restrict__ stats) {
    __shared__ __align__(16) _Float16 sh_cat[TN * 128];
    __shared__ __align__(16) _Float16 sh_h[TN * 128];
    __shared__ __align__(16) _Float16 sh_t[TN * 64];
    __shared__ float sh_red[512];

    int tid = threadIdx.x, lane = tid & 31, w = tid >> 5;
    int r0 = blockIdx.x * TN;

    // coors += cdelta (this block's rows)
    for (int i = tid; i < TN * 3; i += 256) {
        int row = r0 + i / 3;
        if (row < N_NODES) coors[row * 3 + i % 3] += cdelta[row * 3 + i % 3];
    }

    for (int i = tid; i < TN * 128; i += 256) {
        int row = i >> 7, c = i & 127;
        int g = r0 + row;
        float v = 0.0f;
        if (g < N_NODES) v = (c < 64) ? feats[(size_t)g * 64 + c] : m_i[(size_t)g * 64 + (c - 64)];
        sh_cat[i] = (_Float16)v;
    }
    __syncthreads();

    // h = silu(cat @ Wn1 + b)   K=128(4), N=128 (8 tiles)
    {
        const _Float16* Ab = sh_cat + (w & 3) * 16 * 128;
        for (int nt = (w >> 2); nt < 8; nt += 2) {
            v8f acc = {};
#pragma unroll
            for (int kc = 0; kc < 4; ++kc) {
                v16h a = load_A_lds(Ab, 128, lane, kc * 32);
                v16h b = load_B_pk(pWn1, 8, lane, kc, nt);
                acc = wmma_f16(a, b, acc);
            }
            int n = nt * 16 + (lane & 15);
            float bias = bn1[n];
            int mb = (w & 3) * 16 + ((lane >> 4) << 3);
#pragma unroll
            for (int r = 0; r < 8; ++r)
                sh_h[(mb + r) * 128 + n] = (_Float16)silu_f(acc[r] + bias);
        }
    }
    __syncthreads();

    // t = feats + h @ Wn2 + b   K=128(4), N=64 (4 tiles); write feats + sh_t
    {
        const _Float16* Ab = sh_h + (w & 3) * 16 * 128;
        for (int nt = (w >> 2); nt < 4; nt += 2) {
            v8f acc = {};
#pragma unroll
            for (int kc = 0; kc < 4; ++kc) {
                v16h a = load_A_lds(Ab, 128, lane, kc * 32);
                v16h b = load_B_pk(pWn2, 4, lane, kc, nt);
                acc = wmma_f16(a, b, acc);
            }
            int n = nt * 16 + (lane & 15);
            float bias = bn2[n];
            int mb = (w & 3) * 16 + ((lane >> 4) << 3);
#pragma unroll
            for (int r = 0; r < 8; ++r) {
                int g = r0 + mb + r;
                float t = 0.0f;
                if (g < N_NODES) {
                    t = feats[(size_t)g * 64 + n] + acc[r] + bias;
                    feats[(size_t)g * 64 + n] = t;
                }
                sh_t[(mb + r) * 64 + n] = (_Float16)t;
            }
        }
    }
    __syncthreads();

    // u = silu(t @ Wf1 + b)   K=64(2), N=64; LN partial sums
    float s1 = 0.0f, s2 = 0.0f;
    {
        const _Float16* Ab = sh_t + (w & 3) * 16 * 64;
        for (int nt = (w >> 2); nt < 4; nt += 2) {
            v8f acc = {};
#pragma unroll
            for (int kc = 0; kc < 2; ++kc) {
                v16h a = load_A_lds(Ab, 64, lane, kc * 32);
                v16h b = load_B_pk(pWf1, 4, lane, kc, nt);
                acc = wmma_f16(a, b, acc);
            }
            int n = nt * 16 + (lane & 15);
            float bias = bf1[n];
            int mb = (w & 3) * 16 + ((lane >> 4) << 3);
#pragma unroll
            for (int r = 0; r < 8; ++r) {
                int g = r0 + mb + r;
                if (g < N_NODES) {
                    float u = silu_f(acc[r] + bias);
                    u_buf[(size_t)g * 64 + n] = u;
                    s1 += u; s2 += u * u;
                }
            }
        }
    }
    sh_red[tid] = s1; sh_red[256 + tid] = s2;
    __syncthreads();
    for (int off = 128; off > 0; off >>= 1) {
        if (tid < off) {
            sh_red[tid] += sh_red[tid + off];
            sh_red[256 + tid] += sh_red[256 + tid + off];
        }
        __syncthreads();
    }
    if (tid == 0) { atomicAdd(&stats[0], sh_red[0]); atomicAdd(&stats[1], sh_red[256]); }
}

// =====================================================================
// node2: f = graph_ln(u); feats = f @ Wf2 + b
// =====================================================================
__global__ void __launch_bounds__(256)
node2_kernel(const float* __restrict__ u_buf, const float* __restrict__ stats,
             const float* __restrict__ lng, const float* __restrict__ lnb,
             const _Float16* __restrict__ pWf2, const float* __restrict__ bf2,
             float* __restrict__ feats) {
    __shared__ __align__(16) _Float16 sh_f[TN * 64];
    int tid = threadIdx.x, lane = tid & 31, w = tid >> 5;
    int r0 = blockIdx.x * TN;
    float cnt = (float)N_NODES * 64.0f;
    float mu = stats[0] / cnt;
    float var = stats[1] / cnt - mu * mu;
    float inv = rsqrtf(var + 1e-5f);
    for (int i = tid; i < TN * 64; i += 256) {
        int row = i >> 6, c = i & 63;
        int g = r0 + row;
        float v = 0.0f;
        if (g < N_NODES) v = (u_buf[(size_t)g * 64 + c] - mu) * inv * lng[c] + lnb[c];
        sh_f[i] = (_Float16)v;
    }
    __syncthreads();
    const _Float16* Ab = sh_f + (w & 3) * 16 * 64;
    for (int nt = (w >> 2); nt < 4; nt += 2) {
        v8f acc = {};
#pragma unroll
        for (int kc = 0; kc < 2; ++kc) {
            v16h a = load_A_lds(Ab, 64, lane, kc * 32);
            v16h b = load_B_pk(pWf2, 4, lane, kc, nt);
            acc = wmma_f16(a, b, acc);
        }
        int n = nt * 16 + (lane & 15);
        float bias = bf2[n];
        int mb = (w & 3) * 16 + ((lane >> 4) << 3);
#pragma unroll
        for (int r = 0; r < 8; ++r) {
            int g = r0 + mb + r;
            if (g < N_NODES) feats[(size_t)g * 64 + n] = acc[r] + bias;
        }
    }
}

// =====================================================================
// final: out = (feats + ss_embed) @ Wl + bl   [N,20]
// =====================================================================
__global__ void __launch_bounds__(256)
final_kernel(const float* __restrict__ feats, const float* __restrict__ ss_embed,
             const _Float16* __restrict__ pWl, const float* __restrict__ bl,
             float* __restrict__ out) {
    __shared__ __align__(16) _Float16 sh_a[TN * 64];
    int tid = threadIdx.x, lane = tid & 31, w = tid >> 5;
    int r0 = blockIdx.x * TN;
    for (int i = tid; i < TN * 64; i += 256) {
        int row = i >> 6, c = i & 63;
        int g = r0 + row;
        float v = (g < N_NODES) ? feats[(size_t)g * 64 + c] + ss_embed[(size_t)g * 64 + c] : 0.0f;
        sh_a[i] = (_Float16)v;
    }
    __syncthreads();
    const _Float16* Ab = sh_a + (w & 3) * 16 * 64;
    int nt = (w >> 2);  // 0..1 -> N tiles covering 20 cols (padded)
    v8f acc = {};
#pragma unroll
    for (int kc = 0; kc < 2; ++kc) {
        v16h a = load_A_lds(Ab, 64, lane, kc * 32);
        v16h b = load_B_pk(pWl, 2, lane, kc, nt);
        acc = wmma_f16(a, b, acc);
    }
    int n = nt * 16 + (lane & 15);
    int mb = (w & 3) * 16 + ((lane >> 4) << 3);
    if (n < OUTD) {
        float bias = bl[n];
#pragma unroll
        for (int r = 0; r < 8; ++r) {
            int g = r0 + mb + r;
            if (g < N_NODES) out[(size_t)g * OUTD + n] = acc[r] + bias;
        }
    }
}

// =====================================================================
extern "C" void kernel_launch(void* const* d_in, const int* in_sizes, int n_in,
                              void* d_out, int out_size, void* d_ws, size_t ws_size,
                              hipStream_t stream) {
    (void)in_sizes; (void)n_in; (void)out_size; (void)ws_size;
    const float* x         = (const float*)d_in[0];
    const float* pos       = (const float*)d_in[1];
    const float* extra_x   = (const float*)d_in[2];
    const int*   eidx      = (const int*)d_in[3];
    const float* edge_attr = (const float*)d_in[4];
    const float* ss        = (const float*)d_in[5];
    // d_in[6] = batch (unused: single graph)

    // params flattened as jax tree leaves (sorted dict keys):
    // layers[l]: coors1{b,w}, coors2{b,w}, edge1{b,w}, edge2{b,w}, edge_out{b,w},
    //            ff1{b,w}, ff2{b,w}, ln_b, ln_g, node1{b,w}, node2{b,w}  (20 leaves)
    // then lin{b,w}, ss1{b,w}, ss2{b,w}
    const int P = 7;
    auto L = [&](int l, int j) -> const float* { return (const float*)d_in[P + l * 20 + j]; };
    const float* lin_b = (const float*)d_in[P + 40];
    const float* lin_w = (const float*)d_in[P + 41];
    const float* ss1_b = (const float*)d_in[P + 42];
    const float* ss1_w = (const float*)d_in[P + 43];
    const float* ss2_b = (const float*)d_in[P + 44];
    const float* ss2_w = (const float*)d_in[P + 45];

    char* ws = (char*)d_ws;
    size_t off = 0;
    auto alloc = [&](size_t bytes) {
        void* p = ws + off;
        off = (off + bytes + 255) & ~(size_t)255;
        return p;
    };
    float*    feats  = (float*)alloc((size_t)N_NODES * 64 * 4);
    float*    ssem   = (float*)alloc((size_t)N_NODES * 64 * 4);
    float*    coors  = (float*)alloc((size_t)N_NODES * 3 * 4);
    float*    cdelta = (float*)alloc((size_t)N_NODES * 3 * 4);
    float*    m_i    = (float*)alloc((size_t)N_NODES * 64 * 4);
    float*    u_buf  = (float*)alloc((size_t)N_NODES * 64 * 4);
    float*    stats  = (float*)alloc(256);
    _Float16* ea_buf = (_Float16*)alloc((size_t)N_EDGES * 32 * 2);

    // packed f16 weight tiles (per layer + shared)
    _Float16* pk_e1[2]; _Float16* pk_e2[2]; _Float16* pk_c1[2]; _Float16* pk_eo[2];
    _Float16* pk_n1[2]; _Float16* pk_n2[2]; _Float16* pk_f1[2]; _Float16* pk_f2[2];
    for (int l = 0; l < 2; ++l) {
        pk_e1[l] = (_Float16*)alloc((size_t)6 * 21 * 512 * 2);
        pk_e2[l] = (_Float16*)alloc((size_t)11 * 4 * 512 * 2);
        pk_c1[l] = (_Float16*)alloc((size_t)2 * 16 * 512 * 2);
        pk_eo[l] = (_Float16*)alloc((size_t)2 * 2 * 512 * 2);
        pk_n1[l] = (_Float16*)alloc((size_t)4 * 8 * 512 * 2);
        pk_n2[l] = (_Float16*)alloc((size_t)4 * 4 * 512 * 2);
        pk_f1[l] = (_Float16*)alloc((size_t)2 * 4 * 512 * 2);
        pk_f2[l] = (_Float16*)alloc((size_t)2 * 4 * 512 * 2);
    }
    _Float16* pk_lin = (_Float16*)alloc((size_t)2 * 2 * 512 * 2);
    _Float16* pk_ss1 = (_Float16*)alloc((size_t)1 * 4 * 512 * 2);
    _Float16* pk_ss2 = (_Float16*)alloc((size_t)2 * 4 * 512 * 2);

    auto pack = [&](const float* W, _Float16* dst, int K, int N, int KC, int NT) {
        int total = KC * NT * 512;
        pack_w_kernel<<<(total + 255) / 256, 256, 0, stream>>>(W, dst, K, N, KC, NT);
    };
    for (int l = 0; l < 2; ++l) {
        pack(L(l, 5),  pk_e1[l], EIND, EHID, 6, 21);   // edge1.w
        pack(L(l, 7),  pk_e2[l], EHID, MDIMD, 11, 4);  // edge2.w
        pack(L(l, 1),  pk_c1[l], MDIMD, CHID, 2, 16);  // coors1.w
        pack(L(l, 9),  pk_eo[l], MDIMD, EDIMD, 2, 2);  // edge_out.w
        pack(L(l, 17), pk_n1[l], 128, 128, 4, 8);      // node1.w
        pack(L(l, 19), pk_n2[l], 128, 64, 4, 4);       // node2.w
        pack(L(l, 11), pk_f1[l], 64, 64, 2, 4);        // ff1.w
        pack(L(l, 13), pk_f2[l], 64, 64, 2, 4);        // ff2.w
    }
    pack(lin_w, pk_lin, 64, OUTD, 2, 2);
    pack(ss1_w, pk_ss1, 8, 64, 1, 4);
    pack(ss2_w, pk_ss2, 64, 64, 2, 4);

    int gPrep = (N_NODES * 64 + 255) / 256;
    prep_kernel<<<gPrep, 256, 0, stream>>>(x, extra_x, pos, feats, coors);

    int gNode = (N_NODES + TN - 1) / TN;
    ss_kernel<<<gNode, 256, 0, stream>>>(ss, pk_ss1, ss1_b, pk_ss2, ss2_b, ssem);

    int gEdge = N_EDGES / TE;  // 600000/32 = 18750 exact
    for (int l = 0; l < 2; ++l) {
        hipMemsetAsync(m_i, 0, (size_t)N_NODES * 64 * 4, stream);
        hipMemsetAsync(cdelta, 0, (size_t)N_NODES * 3 * 4, stream);
        hipMemsetAsync(stats, 0, 8, stream);
        edge_kernel<<<gEdge, 256, 0, stream>>>(
            feats, coors, eidx,
            (l == 0) ? edge_attr : nullptr, (l == 0) ? nullptr : ea_buf,
            pk_e1[l], L(l, 4),   // edge1
            pk_e2[l], L(l, 6),   // edge2
            pk_c1[l], L(l, 0),   // coors1
            L(l, 3), L(l, 2),    // coors2.w (f32), coors2.b
            pk_eo[l], L(l, 8),   // edge_out
            m_i, cdelta, ea_buf, (l == 0) ? 1 : 0);
        node1_kernel<<<gNode, 256, 0, stream>>>(
            feats, m_i, cdelta, coors,
            pk_n1[l], L(l, 16),  // node1
            pk_n2[l], L(l, 18),  // node2
            pk_f1[l], L(l, 10),  // ff1
            u_buf, stats);
        node2_kernel<<<gNode, 256, 0, stream>>>(
            u_buf, stats,
            L(l, 15), L(l, 14),  // ln_g, ln_b
            pk_f2[l], L(l, 12),  // ff2
            feats);
    }
    final_kernel<<<gNode, 256, 0, stream>>>(feats, ssem, pk_lin, lin_b, (float*)d_out);
}
